// cheb_conv_with_SAt_static_57664230916913
// MI455X (gfx1250) — matrix operand order repaired and checked by reading, not compiled, not run on hardware
//
#include <hip/hip_runtime.h>

typedef __attribute__((ext_vector_type(16))) __bf16       v16bf;
typedef __attribute__((ext_vector_type(8)))  float        v8f;
typedef __attribute__((ext_vector_type(8)))  unsigned int v8u;

union Frag { v8u u; v16bf b; };

#define N_    1000
#define T_    12
#define F_    64
#define KCH   3
#define B_    16
#define BI    64      // i rows per block
#define TT    4       // t's per block
#define NTHR  512     // 16 waves
#define JC    32      // j (contraction) chunk per K-step
#define JSTR  34      // padded j stride in LDS (ushort units) -> 17-bank spread
#define FS    66      // padded f stride in LDS (ushort units) -> 33-bank spread

// round-half-up f32->bf16 (cheap: 2 VALU)
__device__ __forceinline__ unsigned short f2bf(float f) {
    return (unsigned short)((__float_as_uint(f) + 0x8000u) >> 16);
}
// two f32 -> packed bf16x2 (lo in [15:0]): 2 adds + 1 v_perm_b32
__device__ __forceinline__ unsigned int pack2bf(float lo, float hi) {
    unsigned int a = __float_as_uint(lo) + 0x8000u;
    unsigned int b = __float_as_uint(hi) + 0x8000u;
    return __builtin_amdgcn_perm(b, a, 0x07060302u);  // [hi16(b) | hi16(a)]
}

// ---- branch-free global -> LDS staging of one 32-wide j-chunk (j-pairs) ----
template <bool TAIL>
__device__ __forceinline__ void stage_tiles(const float* __restrict__ x,
                                            const float* __restrict__ SAtt,
                                            const float* __restrict__ cheb,
                                            unsigned short* ldsA,
                                            unsigned short* ldsXT,
                                            int b, int i0, int t0, int jc, int tid)
{
    const int bNN = b * (N_ * N_);
    // A tile: A^T[i,j] = SAtt[b,j,i]*cheb[k,j,i] -> bf16 [k][i][j], j-pairs per thread
    #pragma unroll
    for (int r = 0; r < 2; ++r) {
        int idx = r * NTHR + tid;         // 0..1023 = 16 j-pairs x 64 i (i coalesced)
        int jp  = idx >> 6;
        int i   = idx & 63;
        int j   = jp * 2;
        int jg  = jc + j;
        int ig  = i0 + i;
        if (ig > N_ - 1) ig = N_ - 1;     // rows >= N are discarded at store time
        int j0 = jg, j1 = jg + 1;
        if (TAIL) { if (j0 > N_ - 1) j0 = N_ - 1; if (j1 > N_ - 1) j1 = N_ - 1; }
        int off0 = j0 * N_ + ig;
        int off1 = j1 * N_ + ig;
        float sa0 = SAtt[bNN + off0];
        float sa1 = SAtt[bNN + off1];
        if (TAIL) { if (jg     >= N_) sa0 = 0.0f;
                    if (jg + 1 >= N_) sa1 = 0.0f; }
        #pragma unroll
        for (int k = 0; k < KCH; ++k) {
            float c0 = cheb[k * (N_ * N_) + off0];
            float c1 = cheb[k * (N_ * N_) + off1];
            *(unsigned int*)&ldsA[(k * BI + i) * JSTR + j] = pack2bf(sa0 * c0, sa1 * c1);
        }
    }
    // X tile transposed: x[b,t,j,f] -> bf16 [t][f][j], j-pairs per thread
    #pragma unroll
    for (int r = 0; r < 8; ++r) {
        int idx = r * NTHR + tid;         // 0..4095 = 4t x 16 j-pairs x 64f (f coalesced)
        int tt  = idx >> 10;
        int rem = idx & 1023;
        int jp  = rem >> 6;
        int f   = rem & 63;
        int j   = jp * 2;
        int jg  = jc + j;
        int j0 = jg, j1 = jg + 1;
        if (TAIL) { if (j0 > N_ - 1) j0 = N_ - 1; if (j1 > N_ - 1) j1 = N_ - 1; }
        const float* xp = x + ((b * T_ + (t0 + tt)) * N_) * F_ + f;
        float v0 = xp[j0 * F_];
        float v1 = xp[j1 * F_];
        if (TAIL) { if (jg     >= N_) v0 = 0.0f;
                    if (jg + 1 >= N_) v1 = 0.0f; }
        *(unsigned int*)&ldsXT[(tt * F_ + f) * JSTR + j] = pack2bf(v0, v1);
    }
}

// ---- one 32-deep WMMA step: 12 x v_wmma_f32_16x16x32_bf16 per wave ----
__device__ __forceinline__ void wmma_step(const unsigned short* ldsA,
                                          const unsigned short* ldsXT,
                                          v8f acc[KCH][4],
                                          int row, int tw, int mi, int hi)
{
    Frag af[KCH];
    #pragma unroll
    for (int k = 0; k < KCH; ++k)
        #pragma unroll
        for (int v = 0; v < 8; ++v) {
            int jj = ((v < 4) ? 0 : 16) + hi * 8 + (v & 3) * 2;   // 16-bit A layout
            af[k].u[v] = *(const unsigned int*)&ldsA[(k * BI + row * 16 + mi) * JSTR + jj];
        }
    #pragma unroll
    for (int ft = 0; ft < 4; ++ft) {
        Frag bfx;
        #pragma unroll
        for (int v = 0; v < 8; ++v) {
            int jj = ((v < 4) ? 0 : 16) + hi * 8 + (v & 3) * 2;   // 16-bit B layout
            bfx.u[v] = *(const unsigned int*)&ldsXT[(tw * F_ + ft * 16 + mi) * JSTR + jj];
        }
        #pragma unroll
        for (int k = 0; k < KCH; ++k)
            acc[k][ft] = __builtin_amdgcn_wmma_f32_16x16x32_bf16(
                false, af[k].b, false, bfx.b, (short)0, acc[k][ft], false, false);
    }
}

__global__ __launch_bounds__(NTHR)
void cheb_conv_SAt_wmma_kernel(const float* __restrict__ x,
                               const float* __restrict__ SAtt,
                               const float* __restrict__ cheb,
                               const float* __restrict__ Theta,
                               float* __restrict__ out)
{
    // union: stage-1 A (6528) + XT (8704) = 15232  vs  stage-2 Y strips 16*16*FS = 16896
    __shared__ unsigned short lds_u[16 * 16 * FS];       // 33792 B
    __shared__ unsigned short lds_T[KCH * F_ * FS];      // Theta^T [k][o][f], 25344 B

    const int tid  = threadIdx.x;
    const int lane = tid & 31;
    const int wid  = tid >> 5;      // 0..15
    const int row  = wid & 3;       // i strip within block (0..3)
    const int tw   = wid >> 2;      // which t of the quad (0..3)
    const int mi   = lane & 15;
    const int hi   = lane >> 4;     // half-wave selector

    const int b  = blockIdx.z;
    const int i0 = blockIdx.x * BI;
    const int t0 = blockIdx.y * TT;

    // ---- stage Theta once: [k][f][o] f32 -> [k][o][f] bf16 in LDS ----
    #pragma unroll 4
    for (int r = 0; r < 24; ++r) {
        int idx = r * NTHR + tid;           // 0..12287
        int k   = idx >> 12;
        int rem = idx & 4095;
        int f   = rem >> 6;
        int o   = rem & 63;
        lds_T[(k * F_ + o) * FS + f] = f2bf(Theta[(k * F_ + f) * F_ + o]);
    }

    unsigned short* ldsA  = lds_u;                   // [k][i(64)][JSTR]
    unsigned short* ldsXT = lds_u + KCH * BI * JSTR; // [t(4)][f(64)][JSTR]

    const v8f vzero = {0.f,0.f,0.f,0.f,0.f,0.f,0.f,0.f};
    v8f acc[KCH][4];
    #pragma unroll
    for (int k = 0; k < KCH; ++k)
        #pragma unroll
        for (int ft = 0; ft < 4; ++ft) acc[k][ft] = vzero;

    // ============ stage 1: rhs = (cheb*SAtt)^T @ x over j ============
    // 31 unguarded steps (j < 992 < N), then one zero-padded tail step.
    for (int jc = 0; jc < 992; jc += JC) {
        stage_tiles<false>(x, SAtt, cheb, ldsA, ldsXT, b, i0, t0, jc, tid);
        __syncthreads();
        wmma_step(ldsA, ldsXT, acc, row, tw, mi, hi);
        __syncthreads();
    }
    stage_tiles<true>(x, SAtt, cheb, ldsA, ldsXT, b, i0, t0, 992, tid);
    __syncthreads();
    wmma_step(ldsA, ldsXT, acc, row, tw, mi, hi);
    __syncthreads();

    // ============ stage 2: out = relu(sum_k Y_k @ Theta_k) over f ============
    // Wave-private LDS strip (aliases stage-1 region; safe after final barrier).
    unsigned short* ldsY = lds_u + wid * (16 * FS);   // 16 * 16 * 66 = 16896 ushorts total

    v8f oacc[4];
    #pragma unroll
    for (int ot = 0; ot < 4; ++ot) oacc[ot] = vzero;

    #pragma unroll
    for (int k = 0; k < KCH; ++k) {
        // dump this wave's 16x64 Y_k strip (C-layout -> [i][f] bf16)
        #pragma unroll
        for (int ft = 0; ft < 4; ++ft)
            #pragma unroll
            for (int r = 0; r < 8; ++r)
                ldsY[(r + 8 * hi) * FS + ft * 16 + mi] = f2bf(acc[k][ft][r]);
        // contraction over f in two 32-wide steps
        #pragma unroll
        for (int fs = 0; fs < 64; fs += 32) {
            Frag a2;
            #pragma unroll
            for (int v = 0; v < 8; ++v) {
                int ff = fs + ((v < 4) ? 0 : 16) + hi * 8 + (v & 3) * 2;
                a2.u[v] = *(const unsigned int*)&ldsY[mi * FS + ff];
            }
            #pragma unroll
            for (int ot = 0; ot < 4; ++ot) {
                Frag b2;
                #pragma unroll
                for (int v = 0; v < 8; ++v) {
                    int ff = fs + ((v < 4) ? 0 : 16) + hi * 8 + (v & 3) * 2;
                    b2.u[v] = *(const unsigned int*)&lds_T[(k * F_ + ot * 16 + mi) * FS + ff];
                }
                oacc[ot] = __builtin_amdgcn_wmma_f32_16x16x32_bf16(
                    false, a2.b, false, b2.b, (short)0, oacc[ot], false, false);
            }
        }
    }

    // ============ ReLU + coalesced f32 stores ============
    const int tg    = t0 + tw;
    const int ibase = i0 + row * 16 + 8 * hi;
    float* outp = out + ((b * T_ + tg) * N_) * F_ + mi;
    if (i0 + BI <= N_) {
        #pragma unroll
        for (int ot = 0; ot < 4; ++ot)
            #pragma unroll
            for (int r = 0; r < 8; ++r) {
                float v = oacc[ot][r];
                outp[(ibase + r) * F_ + ot * 16] = v > 0.0f ? v : 0.0f;
            }
    } else {
        #pragma unroll
        for (int ot = 0; ot < 4; ++ot)
            #pragma unroll
            for (int r = 0; r < 8; ++r) {
                if (ibase + r < N_) {
                    float v = oacc[ot][r];
                    outp[(ibase + r) * F_ + ot * 16] = v > 0.0f ? v : 0.0f;
                }
            }
    }
}

extern "C" void kernel_launch(void* const* d_in, const int* in_sizes, int n_in,
                              void* d_out, int out_size, void* d_ws, size_t ws_size,
                              hipStream_t stream) {
    (void)in_sizes; (void)n_in; (void)out_size; (void)d_ws; (void)ws_size;
    const float* x     = (const float*)d_in[0];
    const float* SAtt  = (const float*)d_in[1];
    const float* cheb  = (const float*)d_in[2];
    const float* Theta = (const float*)d_in[3];
    float* out = (float*)d_out;
    dim3 grid((N_ + BI - 1) / BI, T_ / TT, B_);   // 16 x 3 x 16 = 768 blocks
    cheb_conv_SAt_wmma_kernel<<<grid, dim3(NTHR), 0, stream>>>(x, SAtt, cheb, Theta, out);
}
